// DANNet2_66099546685596
// MI455X (gfx1250) — compile-verified, matrix operand order checked by reading, not run on hardware
//
#include <hip/hip_runtime.h>
#include <hip/hip_bf16.h>
#include <stdint.h>

#define EPSV 1e-5f

typedef __attribute__((ext_vector_type(16))) _Float16 v16h;
typedef __attribute__((ext_vector_type(8)))  float    v8f;
typedef __attribute__((ext_vector_type(4)))  unsigned v4u;
typedef __attribute__((ext_vector_type(8)))  int      v8i;
typedef __attribute__((ext_vector_type(4)))  int      v4i;

#if __has_builtin(__builtin_amdgcn_tensor_load_to_lds)
#define HAVE_TDM 1
#else
#define HAVE_TDM 0
#endif

__device__ __forceinline__ unsigned pack2(float a, float b) {
  union { _Float16 h[2]; unsigned u; } x;
  x.h[0] = (_Float16)a; x.h[1] = (_Float16)b;
  return x.u;
}

union FragU { unsigned u[8]; v16h h; };

// -------------------------------------------------------------------------
// WMMA GEMM: Out[M,N](f16) = epilogue(A[M,K](f16) @ W[K,N](f32))
//   mode 0: out = y + bias[n]                               (pre-LN raw)
//   mode 1: out = relu(y*s + t), s=g*rsqrt(v+eps), t=(bias-mean)*s+beta
// Tile 128x128 per 256-thread block, BK=32, software-pipelined with
// double-buffered LDS. A tiles move via the Tensor Data Mover (async,
// issued one tile ahead, waited at end of the compute iteration); the TDM
// pad feature (16 dwords -> +1 dword) reproduces the [128][17] layout.
// blockIdx.z selects the expert via per-z strides.
// -------------------------------------------------------------------------
__global__ __launch_bounds__(256) void gemm_wmma_kernel(
    const _Float16* __restrict__ A, long long sA,
    const float* __restrict__ W, long long sW,
    _Float16* __restrict__ Out, long long sO,
    int M, int N, int K,
    const float* __restrict__ bias,
    const float* __restrict__ gam, const float* __restrict__ bet,
    const float* __restrict__ mean, const float* __restrict__ var,
    long long pstride, int mode)
{
  __shared__ unsigned Als[2][128][17];  // f16 pairs: [m][kpair], +1 pad dword
  __shared__ unsigned Bls[2][128][17];  // f16 pairs: [n][kpair] (W transposed)

  const int tid  = threadIdx.x;
  const int lane = tid & 31;
  const int wave = tid >> 5;
  const int wm   = wave & 3;    // 4 wave-rows of 32
  const int wn   = wave >> 2;   // 2 wave-cols of 64
  const int lrow = lane & 15;
  const int hi   = lane >> 4;
  const int m0   = blockIdx.x * 128;
  const int n0   = blockIdx.y * 128;
  const int z    = blockIdx.z;

  const _Float16* Az = A + (size_t)z * sA;
  const float*    Wz = W + (size_t)z * sW;
  _Float16*       Oz = Out + (size_t)z * sO;
  const float*    biasz = bias + (size_t)z * pstride;

  // B staging roles: n = tid&127, k half = tid>>7 (16 k's each)
  const int bnc = tid & 127;
  const int bkg = tid >> 7;

  v8f acc[2][4] = {};

#if HAVE_TDM
  // constant part of the Tensor DMA descriptor (group 1)
  const unsigned td0 = 0x7fffffffu, td1 = 0x7fffffffu;
  v8i g1;
  g1[0] = (2 << 16)          // data_size = 4B
        | (1 << 20)          // pad_enable
        | (3 << 22);         // pad_interval: 16 dwords (pad_amount 0 -> 1 dw)
  g1[1] = (int)((td0 & 0xFFFFu) << 16);                       // tensor_dim0 lo
  g1[2] = (int)(((td0 >> 16) & 0xFFFFu) | ((td1 & 0xFFFFu) << 16));
  g1[3] = (int)(((td1 >> 16) & 0xFFFFu) | (16u << 16));       // tile_dim0 = 16 dw
  g1[4] = 128;                                                // tile_dim1 = 128
  g1[5] = K >> 1;                                             // dim0 stride (dw)
  g1[6] = 0; g1[7] = 0;
  const v4i z4 = {0, 0, 0, 0};
#if __clang_major__ >= 23
  const v8i z8 = {0, 0, 0, 0, 0, 0, 0, 0};
#endif
  const unsigned ldsA0 = (unsigned)(size_t)(void*)&Als[0][0][0];
  const unsigned ldsA1 = (unsigned)(size_t)(void*)&Als[1][0][0];
#endif

  // ---- staging helpers ----
  auto stageA = [&](int kt, int buf) {   // async TDM issue (wave 0 only)
#if HAVE_TDM
    if (wave == 0) {
      const unsigned long long ga =
          (unsigned long long)(size_t)(Az + (size_t)m0 * K + (kt << 5));
      v4u g0;
      g0.x = 1u;                                   // count = 1 valid descriptor
      g0.y = buf ? ldsA1 : ldsA0;                  // LDS destination
      g0.z = (unsigned)(ga & 0xFFFFFFFFu);         // global addr lo
      g0.w = (unsigned)(ga >> 32) | (2u << 30);    // global addr hi | type=2
#if __clang_major__ >= 23
      __builtin_amdgcn_tensor_load_to_lds(g0, g1, z4, z4, z8, 0);
#else
      __builtin_amdgcn_tensor_load_to_lds(g0, g1, z4, z4, 0);
#endif
    }
#else
    {  // manual A staging fallback: 128x32 f16, uint4 (8 halves) per access
      const int r0 = tid >> 2, part = tid & 3;
      #pragma unroll
      for (int p = 0; p < 2; ++p) {
        const int r = r0 + p * 64;
        const uint4 v =
            *((const uint4*)(Az + (size_t)(m0 + r) * K + (kt << 5)) + part);
        Als[buf][r][part * 4 + 0] = v.x;
        Als[buf][r][part * 4 + 1] = v.y;
        Als[buf][r][part * 4 + 2] = v.z;
        Als[buf][r][part * 4 + 3] = v.w;
      }
    }
#endif
  };
  auto waitA = [&]() {                   // retire outstanding TDM (wave 0)
#if HAVE_TDM
    if (wave == 0) __builtin_amdgcn_s_wait_tensorcnt((short)0);
#endif
  };
  auto stageB = [&](int kt, int buf) {   // transpose + f32->f16
    const float* wp = Wz + (size_t)((kt << 5) + bkg * 16) * N + n0 + bnc;
    #pragma unroll
    for (int i = 0; i < 8; ++i) {
      const float e0 = wp[(size_t)(2 * i) * N];
      const float e1 = wp[(size_t)(2 * i + 1) * N];
      Bls[buf][bnc][bkg * 8 + i] = pack2(e0, e1);
    }
  };

  const int ktiles = K >> 5;

  // ---- prologue: stage tile 0 into buffer 0 ----
  stageA(0, 0);
  stageB(0, 0);
  waitA();
  __syncthreads();

  // ---- pipelined main loop ----
  for (int kt = 0; kt < ktiles; ++kt) {
    const int cur = kt & 1, nxt = cur ^ 1;
    const bool more = (kt + 1 < ktiles);

    if (more) {                 // stage next tile while computing this one
      stageA(kt + 1, nxt);      // async DMA, no wait here
      stageB(kt + 1, nxt);
      if (kt + 2 < ktiles)
        __builtin_prefetch(Wz + (size_t)((kt + 2) << 5) * N + n0 + bnc, 0, 1);
    }

    // fragment loads per ISA 16-bit A 16x32 layout
    FragU a[2], b[4];
    #pragma unroll
    for (int j = 0; j < 8; ++j) {
      const int kp = ((j & 4) << 1) + (hi << 2) + (j & 3);
      #pragma unroll
      for (int mi = 0; mi < 2; ++mi)
        a[mi].u[j] = Als[cur][wm * 32 + mi * 16 + lrow][kp];
      #pragma unroll
      for (int ni = 0; ni < 4; ++ni)
        b[ni].u[j] = Bls[cur][wn * 64 + ni * 16 + lrow][kp];
    }
    #pragma unroll
    for (int mi = 0; mi < 2; ++mi)
      #pragma unroll
      for (int ni = 0; ni < 4; ++ni)
        acc[mi][ni] = __builtin_amdgcn_wmma_f32_16x16x32_f16(
            false, a[mi].h, false, b[ni].h, (short)0, acc[mi][ni], false, false);

    if (more) {
      waitA();                  // DMA had a full tile of WMMA time to finish
      __syncthreads();          // publish next tile; retire reads of cur
    }
  }

  // ---- epilogue (C layout: n = lane%16, m = r + 8*hi) ----
  #pragma unroll
  for (int mi = 0; mi < 2; ++mi) {
    #pragma unroll
    for (int ni = 0; ni < 4; ++ni) {
      const int n = n0 + wn * 64 + ni * 16 + lrow;
      float s, t;
      if (mode == 1) {
        s = gam[n] * rsqrtf(var[n] + EPSV);
        t = (biasz[n] - mean[n]) * s + bet[n];
      } else {
        s = 1.0f; t = biasz[n];
      }
      #pragma unroll
      for (int r = 0; r < 8; ++r) {
        const int m = m0 + wm * 32 + mi * 16 + hi * 8 + r;
        float y = fmaf(acc[mi][ni][r], s, t);
        if (mode == 1) y = fmaxf(y, 0.0f);
        Oz[(size_t)m * N + n] = (_Float16)y;
      }
    }
  }
}

// ---- fp32 -> f16 convert ----
__global__ void f32_to_f16_kernel(const float* __restrict__ in,
                                  _Float16* __restrict__ out, int n)
{
  const int i = (blockIdx.x * blockDim.x + threadIdx.x) * 4;
  if (i + 3 < n) {
    const float4 f = *(const float4*)(in + i);
    out[i] = (_Float16)f.x; out[i + 1] = (_Float16)f.y;
    out[i + 2] = (_Float16)f.z; out[i + 3] = (_Float16)f.w;
  }
}

// ---- LayerNorm + ReLU over last dim (f16 in/out), block per row ----
__global__ __launch_bounds__(256) void ln_relu_kernel(
    const _Float16* __restrict__ Y, long long yEs,
    _Float16* __restrict__ Out, long long oEs,
    const float* __restrict__ g, const float* __restrict__ be, long long gEs,
    int N, int rowsPerE)
{
  __shared__ float s1[256], s2[256];
  const int row = blockIdx.x, tid = threadIdx.x;
  const int e = row / rowsPerE, r = row % rowsPerE;
  const _Float16* y = Y + (size_t)e * yEs + (size_t)r * N;
  _Float16* o = Out + (size_t)e * oEs + (size_t)r * N;
  const float* gg = g + (size_t)e * gEs;
  const float* bb = be + (size_t)e * gEs;
  float sum = 0.f, sq = 0.f;
  for (int i = tid; i < N; i += 256) { const float v = (float)y[i]; sum += v; sq += v * v; }
  s1[tid] = sum; s2[tid] = sq; __syncthreads();
  for (int s = 128; s > 0; s >>= 1) {
    if (tid < s) { s1[tid] += s1[tid + s]; s2[tid] += s2[tid + s]; }
    __syncthreads();
  }
  const float mu = s1[0] / (float)N;
  const float var = s2[0] / (float)N - mu * mu;
  const float rs = rsqrtf(var + EPSV);
  for (int i = tid; i < N; i += 256)
    o[i] = (_Float16)fmaxf(((float)y[i] - mu) * rs * gg[i] + bb[i], 0.0f);
}

// ---- small dense head over f16 activations; thread per (e,b,n) ----
__global__ void dense_head_kernel(
    const _Float16* __restrict__ X, long long xEs,
    const float* __restrict__ W, long long wEs,
    const float* __restrict__ bias, long long bEs,
    float* __restrict__ Out, int Brows, int N, int K, int ostride, int oEoff,
    int total)
{
  const int idx = blockIdx.x * blockDim.x + threadIdx.x;
  if (idx >= total) return;
  const int e = idx / (Brows * N);
  const int rem = idx % (Brows * N);
  const int b = rem / N, n = rem % N;
  const _Float16* x = X + (size_t)e * xEs + (size_t)b * K;
  const float* w = W + (size_t)e * wEs;
  float s = bias[(size_t)e * bEs + n];
  for (int k = 0; k < K; ++k) s = fmaf((float)x[k], w[(size_t)k * N + n], s);
  Out[(size_t)b * ostride + e * oEoff + n] = s;
}

// ---- log_softmax over 8 logits + argmax routing label ----
__global__ void src_softmax_kernel(const float* __restrict__ logits,
                                   float* __restrict__ out, int* __restrict__ labels)
{
  const int b = blockIdx.x * blockDim.x + threadIdx.x;
  if (b >= 2048) return;
  const float* l = logits + b * 8;
  float m = l[0]; int am = 0;
  #pragma unroll
  for (int i = 1; i < 8; ++i) if (l[i] > m) { m = l[i]; am = i; }
  float s = 0.f;
  #pragma unroll
  for (int i = 0; i < 8; ++i) s += expf(l[i] - m);
  const float lse = logf(s);
  #pragma unroll
  for (int i = 0; i < 8; ++i) out[b * 8 + i] = l[i] - m - lse;
  labels[b] = am;
}

// ---- log_softmax over 2 logits ----
__global__ void logsoftmax2_kernel(const float* __restrict__ logits,
                                   float* __restrict__ out)
{
  const int b = blockIdx.x * blockDim.x + threadIdx.x;
  if (b >= 2048) return;
  const float a = logits[b * 2], c = logits[b * 2 + 1];
  const float m = fmaxf(a, c);
  const float lse = m + logf(expf(a - m) + expf(c - m));
  out[b * 2] = a - lse; out[b * 2 + 1] = c - lse;
}

// ---- gather routed expert + log_softmax(2) ----
__global__ void gather_sout_kernel(const float* __restrict__ eo,
                                   const int* __restrict__ labels,
                                   float* __restrict__ out)
{
  const int b = blockIdx.x * blockDim.x + threadIdx.x;
  if (b >= 2048) return;
  const int lab = labels[b];
  const float a = eo[b * 16 + lab * 2], c = eo[b * 16 + lab * 2 + 1];
  const float m = fmaxf(a, c);
  const float lse = m + logf(expf(a - m) + expf(c - m));
  out[b * 2] = a - lse; out[b * 2 + 1] = c - lse;
}

// ---- share passthrough copy (float4) ----
__global__ void copy4_kernel(const float4* __restrict__ in,
                             float4* __restrict__ out, int n4)
{
  const int i = blockIdx.x * blockDim.x + threadIdx.x;
  if (i < n4) out[i] = in[i];
}

extern "C" void kernel_launch(void* const* d_in, const int* in_sizes, int n_in,
                              void* d_out, int out_size, void* d_ws, size_t ws_size,
                              hipStream_t stream)
{
  (void)in_sizes; (void)n_in; (void)out_size; (void)ws_size;
  const int B = 2048, D = 1280;
  const long long ZN = (long long)B * 1024;   // per-expert elements (wide layers)
  const long long ZH = (long long)B * 512;    // per-expert elements (512 layers)

  const float* src  = (const float*)d_in[0];
  const float* bw1  = (const float*)d_in[1];
  const float* bb1  = (const float*)d_in[2];
  const float* bg1  = (const float*)d_in[3];
  const float* bbe1 = (const float*)d_in[4];
  const float* bm1  = (const float*)d_in[5];
  const float* bv1  = (const float*)d_in[6];
  const float* bw2  = (const float*)d_in[7];
  const float* bb2  = (const float*)d_in[8];
  const float* bg2  = (const float*)d_in[9];
  const float* bbe2 = (const float*)d_in[10];
  const float* bm2  = (const float*)d_in[11];
  const float* bv2  = (const float*)d_in[12];
  const float* cw   = (const float*)d_in[13];
  const float* cb   = (const float*)d_in[14];
  const float* dw1  = (const float*)d_in[15];
  const float* db1  = (const float*)d_in[16];
  const float* dg1  = (const float*)d_in[17];
  const float* dbe1 = (const float*)d_in[18];
  const float* dm1  = (const float*)d_in[19];
  const float* dv1  = (const float*)d_in[20];
  const float* dw2  = (const float*)d_in[21];
  const float* db2  = (const float*)d_in[22];
  const float* dg2  = (const float*)d_in[23];
  const float* dbe2 = (const float*)d_in[24];
  const float* dm2  = (const float*)d_in[25];
  const float* dv2  = (const float*)d_in[26];
  const float* dw3  = (const float*)d_in[27];
  const float* db3  = (const float*)d_in[28];
  const float* dg3  = (const float*)d_in[29];
  const float* dbe3 = (const float*)d_in[30];
  const float* dm3  = (const float*)d_in[31];
  const float* dv3  = (const float*)d_in[32];
  const float* dw4  = (const float*)d_in[33];
  const float* db4  = (const float*)d_in[34];
  const float* ew1  = (const float*)d_in[35];
  const float* eb1  = (const float*)d_in[36];
  const float* eg1  = (const float*)d_in[37];
  const float* ebe1 = (const float*)d_in[38];
  const float* ew2  = (const float*)d_in[39];
  const float* eb2  = (const float*)d_in[40];
  const float* eg2  = (const float*)d_in[41];
  const float* ebe2 = (const float*)d_in[42];
  const float* ew3  = (const float*)d_in[43];
  const float* eb3  = (const float*)d_in[44];
  const float* eg3  = (const float*)d_in[45];
  const float* ebe3 = (const float*)d_in[46];
  const float* ew4  = (const float*)d_in[47];
  const float* eb4  = (const float*)d_in[48];

  // workspace layout: f16 region first (16B-aligned chunks), fp32 after
  _Float16* h      = (_Float16*)d_ws;
  _Float16* srcH   = h;                          // [B,1280]
  _Float16* act1h  = srcH + (size_t)B * 1280;    // [B,512]
  _Float16* act2h  = act1h + (size_t)B * 512;    // [B,256]
  _Float16* dact1h = act2h + (size_t)B * 256;    // [B,1024]
  _Float16* dact2h = dact1h + (size_t)B * 1024;  // [B,1024]
  _Float16* dact3h = dact2h + (size_t)B * 1024;  // [B,512]
  _Float16* pingA  = dact3h + (size_t)B * 512;   // [8,B,1024]
  _Float16* pingB  = pingA + (size_t)8 * ZN;     // [8,B,1024]
  float* logits8   = (float*)(pingB + (size_t)8 * ZN);  // [B,8]
  float* dlog      = logits8 + (size_t)B * 8;    // [B,2]
  float* eo        = dlog + (size_t)B * 2;       // [B,8,2]
  int*   labels    = (int*)(eo + (size_t)B * 16);

  // output: src[B,8] | dclf[B,2] | sout[B,2] | share[B,1280]
  float* out_src   = (float*)d_out;
  float* out_dclf  = out_src + (size_t)B * 8;
  float* out_sout  = out_dclf + (size_t)B * 2;
  float* out_share = out_sout + (size_t)B * 2;

  const dim3 blk(256);

  // ---- feature convert ----
  f32_to_f16_kernel<<<(B * D / 4 + 255) / 256, blk, 0, stream>>>(src, srcH, B * D);

  // ---- bottleneck + classifier head ----
  gemm_wmma_kernel<<<dim3(16, 4, 1), blk, 0, stream>>>(
      srcH, 0, bw1, 0, act1h, 0, B, 512, D, bb1, bg1, bbe1, bm1, bv1, 0, 1);
  gemm_wmma_kernel<<<dim3(16, 2, 1), blk, 0, stream>>>(
      act1h, 0, bw2, 0, act2h, 0, B, 256, 512, bb2, bg2, bbe2, bm2, bv2, 0, 1);
  dense_head_kernel<<<(B * 8 + 255) / 256, blk, 0, stream>>>(
      act2h, 0, cw, 0, cb, 0, logits8, B, 8, 256, 8, 0, B * 8);
  src_softmax_kernel<<<8, 256, 0, stream>>>(logits8, out_src, labels);

  // ---- domain classifier (grad-reverse is identity in forward) ----
  gemm_wmma_kernel<<<dim3(16, 8, 1), blk, 0, stream>>>(
      srcH, 0, dw1, 0, dact1h, 0, B, 1024, D, db1, dg1, dbe1, dm1, dv1, 0, 1);
  gemm_wmma_kernel<<<dim3(16, 8, 1), blk, 0, stream>>>(
      dact1h, 0, dw2, 0, dact2h, 0, B, 1024, 1024, db2, dg2, dbe2, dm2, dv2, 0, 1);
  gemm_wmma_kernel<<<dim3(16, 4, 1), blk, 0, stream>>>(
      dact2h, 0, dw3, 0, dact3h, 0, B, 512, 1024, db3, dg3, dbe3, dm3, dv3, 0, 1);
  dense_head_kernel<<<(B * 2 + 255) / 256, blk, 0, stream>>>(
      dact3h, 0, dw4, 0, db4, 0, dlog, B, 2, 512, 2, 0, B * 2);
  logsoftmax2_kernel<<<8, 256, 0, stream>>>(dlog, out_dclf);

  // ---- 8 expert MLPs, batched over gridDim.z ----
  gemm_wmma_kernel<<<dim3(16, 8, 8), blk, 0, stream>>>(
      srcH, 0, ew1, (long long)D * 1024, pingA, ZN, B, 1024, D,
      eb1, nullptr, nullptr, nullptr, nullptr, 1024, 0);
  ln_relu_kernel<<<8 * B, blk, 0, stream>>>(pingA, ZN, pingB, ZN, eg1, ebe1, 1024,
                                            1024, B);
  gemm_wmma_kernel<<<dim3(16, 8, 8), blk, 0, stream>>>(
      pingB, ZN, ew2, 1024LL * 1024, pingA, ZN, B, 1024, 1024,
      eb2, nullptr, nullptr, nullptr, nullptr, 1024, 0);
  ln_relu_kernel<<<8 * B, blk, 0, stream>>>(pingA, ZN, pingB, ZN, eg2, ebe2, 1024,
                                            1024, B);
  gemm_wmma_kernel<<<dim3(16, 4, 8), blk, 0, stream>>>(
      pingB, ZN, ew3, 1024LL * 512, pingA, ZH, B, 512, 1024,
      eb3, nullptr, nullptr, nullptr, nullptr, 512, 0);
  ln_relu_kernel<<<8 * B, blk, 0, stream>>>(pingA, ZH, pingB, ZH, eg3, ebe3, 512,
                                            512, B);
  dense_head_kernel<<<(8 * B * 2 + 255) / 256, blk, 0, stream>>>(
      pingB, ZH, ew4, 1024, eb4, 2, eo, B, 2, 512, 16, 2, 8 * B * 2);

  // ---- gather routed expert output + share passthrough ----
  gather_sout_kernel<<<8, 256, 0, stream>>>(eo, labels, out_sout);
  copy4_kernel<<<(B * D / 4 + 255) / 256, blk, 0, stream>>>(
      (const float4*)src, (float4*)out_share, B * D / 4);
}
